// StyleConv_56590489092563
// MI455X (gfx1250) — compile-verified, matrix operand order checked by reading, not run on hardware
//
#include <hip/hip_runtime.h>
#include <hip/hip_bf16.h>
#include <math.h>

typedef __bf16 bf16;
typedef __attribute__((ext_vector_type(8)))  __bf16 v8bf;
typedef __attribute__((ext_vector_type(16))) __bf16 v16bf;
typedef __attribute__((ext_vector_type(8)))  float  v8f;
typedef __attribute__((ext_vector_type(4)))  int    v4i;

#define AS1 __attribute__((address_space(1)))
#define AS3 __attribute__((address_space(3)))

#define BATCH 8
#define CIN   256
#define COUT  512
#define HW    64
#define HP    65     // blurred size
#define OHW   32     // conv output size
#define NSF   512

#define APITCH 40            // bf16 elements per LDS row (32 data + 8 pad, 80B)
#define TSTRIDE (128 * APITCH)  // one A or B buffer (128 rows)

// ---------------------------------------------------------------------------
// CDNA5 async global->LDS copy (ASYNCcnt-tracked) + waits
// ---------------------------------------------------------------------------
__device__ __forceinline__ void async_copy_b128(const bf16* g, bf16* l) {
#if __has_builtin(__builtin_amdgcn_global_load_async_to_lds_b128)
    __builtin_amdgcn_global_load_async_to_lds_b128(
        (AS1 v4i*)g, (AS3 v4i*)l, 0, 0);
#else
    unsigned l32 = (unsigned)(size_t)(AS3 void*)l;
    asm volatile("global_load_async_to_lds_b128 %0, %1, off"
                 :: "v"(l32), "v"(g) : "memory");
#endif
}

__device__ __forceinline__ void wait_async_4() {
#if __has_builtin(__builtin_amdgcn_s_wait_asynccnt)
    __builtin_amdgcn_s_wait_asynccnt(4);
#else
    asm volatile("s_wait_asynccnt 0x4" ::: "memory");
#endif
}

__device__ __forceinline__ void wait_async_0() {
#if __has_builtin(__builtin_amdgcn_s_wait_asynccnt)
    __builtin_amdgcn_s_wait_asynccnt(0);
#else
    asm volatile("s_wait_asynccnt 0x0" ::: "memory");
#endif
}

// Build a 16-element bf16 fragment from the two 16B K-chunks this lane owns.
__device__ __forceinline__ v16bf ld_frag(const bf16* p) {
    v8bf lo = *(const v8bf*)p;
    v8bf hi = *(const v8bf*)(p + 16);
    return __builtin_shufflevector(lo, hi,
        0,1,2,3,4,5,6,7,8,9,10,11,12,13,14,15);
}

// ---------------------------------------------------------------------------
// Kernel 1: s[b,c] = style[b,:] . (w_mod[c,:] / sqrt(512)) + b_mod[c]
// ---------------------------------------------------------------------------
__global__ void style_mod_kernel(const float* __restrict__ style,
                                 const float* __restrict__ w_mod,
                                 const float* __restrict__ b_mod,
                                 float* __restrict__ s) {
    int b = blockIdx.x;
    int c = threadIdx.x;
    __shared__ float st[NSF];
    for (int k = threadIdx.x; k < NSF; k += 256) st[k] = style[b * NSF + k];
    __syncthreads();
    const float* wr = w_mod + (size_t)c * NSF;
    float acc = 0.f;
    for (int k = 0; k < NSF; ++k) acc += st[k] * wr[k];
    s[b * CIN + c] = acc * 0.04419417382415922f /* 1/sqrt(512) */ + b_mod[c];
}

// ---------------------------------------------------------------------------
// Kernel 2: demod[b,co] = rsqrt( (1/2304) * sum_c s[b,c]^2 * sum_kk w^2 + eps )
// ---------------------------------------------------------------------------
__global__ void demod_kernel(const float* __restrict__ conv_w,
                             const float* __restrict__ s,
                             float* __restrict__ demod) {
    int co = blockIdx.x;
    int b  = blockIdx.y;
    int c  = threadIdx.x;
    float sv = s[b * CIN + c];
    const float* wp = conv_w + ((size_t)co * CIN + c) * 9;
    float acc = 0.f;
#pragma unroll
    for (int t = 0; t < 9; ++t) { float w = wp[t]; acc += w * w; }
    acc *= sv * sv;
    __shared__ float red[256];
    red[c] = acc;
    __syncthreads();
    for (int off = 128; off > 0; off >>= 1) {
        if (c < off) red[c] += red[c + off];
        __syncthreads();
    }
    if (c == 0)
        demod[b * COUT + co] = rsqrtf(red[0] * (1.f / 2304.f) + 1e-8f);
}

// ---------------------------------------------------------------------------
// Kernel 3: pack conv_w [co][c][kh][kw] f32 -> wt [t][co][c] bf16  (t=kh*3+kw)
// ---------------------------------------------------------------------------
__global__ void pack_w_kernel(const float* __restrict__ conv_w,
                              bf16* __restrict__ wt) {
    int g = blockIdx.x * 256 + threadIdx.x;   // g = co*256 + c
    int co = g >> 8;
    int c  = g & 255;
    const float* wp = conv_w + (size_t)g * 9;
#pragma unroll
    for (int t = 0; t < 9; ++t)
        wt[((size_t)(t * COUT + co)) * CIN + c] = (bf16)wp[t];
}

// ---------------------------------------------------------------------------
// Kernel 4: FIR 4x4 blur (pad 2) + modulate by s[b,c]/48 + pack bf16 NHWC
// ---------------------------------------------------------------------------
__global__ void __launch_bounds__(256)
fir_mod_pack_kernel(const float* __restrict__ x,
                    const float* __restrict__ s,
                    bf16* __restrict__ xmod) {
    int id = blockIdx.x;
    int cc = id & 7;  id >>= 3;          // channel chunk 0..7
    int h  = id % HP;                    // output row 0..64
    int b  = id / HP;
    __shared__ float tile[4 * 64 * 33];  // [i][w][c], c padded to 33
    int tid = threadIdx.x;
    int wld = tid & 63;
    int ild = tid >> 6;                  // 0..3
    int r   = h - 2 + ild;
    for (int c = 0; c < 32; ++c) {
        float v = 0.f;
        if (r >= 0 && r < HW)
            v = x[(((size_t)(b * CIN + cc * 32 + c)) * HW + r) * HW + wld];
        tile[(ild * 64 + wld) * 33 + c] = v;
    }
    __syncthreads();

    const float k1[4] = {1.f, 3.f, 3.f, 1.f};
    for (int o = tid; o < 65 * 32; o += 256) {
        int c = o & 31;
        int w = o >> 5;                  // 0..64
        float acc = 0.f;
#pragma unroll
        for (int i = 0; i < 4; ++i) {
#pragma unroll
            for (int j = 0; j < 4; ++j) {
                int col = w - 2 + j;
                if (col >= 0 && col < 64)
                    acc += k1[i] * k1[j] * tile[(i * 64 + col) * 33 + c];
            }
        }
        float sc = s[b * CIN + cc * 32 + c] * (1.f / 48.f); // 1/sqrt(2304)
        acc *= (1.f / 64.f) * sc;                           // FIR normalization
        xmod[(((size_t)(b * HP + h)) * HP + w) * CIN + cc * 32 + c] = (bf16)acc;
    }
}

// ---------------------------------------------------------------------------
// Kernel 5: implicit GEMM, triple-buffered async-LDS, 1 barrier per K-step
//   M = (b,oh,ow)=8192, N = co=512, K = (kh,kw,c)=2304
//   Block tile M128 x N128; 8 waves; wave tile 32M x 64N (8 accumulators).
//   Per K-step every thread issues exactly 4 async b128 copies (A+B = 16KB).
// ---------------------------------------------------------------------------
__global__ void __launch_bounds__(256)
modconv_wmma_kernel(const bf16* __restrict__ xmod,
                    const bf16* __restrict__ wt,
                    const float* __restrict__ demod,
                    const float* __restrict__ noise,
                    const float* __restrict__ nwp,
                    const float* __restrict__ act_bias,
                    float* __restrict__ out) {
    __shared__ __align__(16) bf16 sA[3][TSTRIDE];   // 30 KB
    __shared__ __align__(16) bf16 sB[3][TSTRIDE];   // 30 KB

    const int tm = blockIdx.x;            // 0..63 : (b, 4-row group)
    const int tn = blockIdx.y;            // 0..3  : 128-wide co panel
    const int b   = tm >> 3;
    const int oh0 = (tm & 7) * 4;
    const int coPanel = tn * 128;

    const int tid  = threadIdx.x;
    const int lane = tid & 31;
    const int wv   = tid >> 5;
    const int wave_m = wv >> 1;           // 0..3 : which output row
    const int wave_n = wv & 1;            // 0..1 : which 64-co slice
    const int nm   = lane & 15;
    const int koff = (lane & 16) ? 8 : 0; // ISA K-chunk ownership per lane

    // --- async load assignment: 4 x b128 per thread per K-step (uniform) ---
    const int rowA  = tid >> 1;           // 0..127 (m row / B co row)
    const int cpair = (tid & 1) * 16;     // bf16 offset of 32B half-row
    const int owA   = rowA & 31;
    const int ohA   = oh0 + (rowA >> 5);
    const size_t aRowBase = ((size_t)(b * HP + 2 * ohA)) * HP + 2 * owA;

    bf16* lA = &sA[0][rowA * APITCH + cpair];
    bf16* lB = &sB[0][rowA * APITCH + cpair];

    auto issue = [&](int s, int buf) {
        int t  = s >> 3;                  // 0..8 : (kh,kw)
        int ck = s & 7;                   // 32-channel chunk
        int kh = (t * 11) >> 5;           // t/3
        int kw = t - kh * 3;
        const bf16* ga = xmod + (aRowBase + kh * HP + kw) * CIN + ck * 32 + cpair;
        const bf16* gb = wt + ((size_t)(t * COUT + coPanel + rowA)) * CIN + ck * 32 + cpair;
        bf16* la = lA + buf * TSTRIDE;
        bf16* lb = lB + buf * TSTRIDE;
        async_copy_b128(ga,     la);
        async_copy_b128(ga + 8, la + 8);
        async_copy_b128(gb,     lb);
        async_copy_b128(gb + 8, lb + 8);
    };

    v8f acc[2][4] = {};

    issue(0, 0);
    int buf = 0;
    for (int s = 0; s < 72; ++s) {
        int nbuf = (buf == 2) ? 0 : buf + 1;
        if (s + 1 < 72) { issue(s + 1, nbuf); wait_async_4(); }
        else            { wait_async_0(); }
        __syncthreads();   // barrier(s): step-s tiles visible; also proves
                           // all waves finished compute(s-2) -> nbuf reuse safe

        const bf16* pa = &sA[0][buf * TSTRIDE + (wave_m * 32 + nm) * APITCH + koff];
        v16bf A0 = ld_frag(pa);
        v16bf A1 = ld_frag(pa + 16 * APITCH);
#pragma unroll
        for (int ni = 0; ni < 4; ++ni) {
            const bf16* pb = &sB[0][buf * TSTRIDE +
                                    (wave_n * 64 + ni * 16 + nm) * APITCH + koff];
            v16bf Bf = ld_frag(pb);
            acc[0][ni] = __builtin_amdgcn_wmma_f32_16x16x32_bf16(
                false, A0, false, Bf, (short)0, acc[0][ni], false, false);
            acc[1][ni] = __builtin_amdgcn_wmma_f32_16x16x32_bf16(
                false, A1, false, Bf, (short)0, acc[1][ni], false, false);
        }
        buf = nbuf;
    }

    // --- epilogue: demod + noise + bias + leaky-relu(0.2)*sqrt(2) ----------
    // D layout: VGPR r -> M = r + (lane>=16 ? 8 : 0), N = lane&15
    const int oh     = oh0 + wave_m;
    const float nw   = nwp[0];
    const int owHalf = (lane & 16) ? 8 : 0;

#define LREL(v) ((((v) >= 0.f) ? (v) : 0.2f * (v)) * 1.41421356237f)
#pragma unroll
    for (int mi = 0; mi < 2; ++mi) {
        const int owb = mi * 16 + owHalf;
        const float* np = noise + ((size_t)(b * OHW + oh)) * OHW + owb;
        float nv[8];
#pragma unroll
        for (int r2 = 0; r2 < 8; ++r2) nv[r2] = nw * np[r2];
#pragma unroll
        for (int ni = 0; ni < 4; ++ni) {
            const int co0 = coPanel + wave_n * 64 + ni * 16 + nm;
            const float dm   = demod[b * COUT + co0];
            const float bias = act_bias[co0];
            const v8f a = acc[mi][ni];
            float* op = out + (((size_t)(b * COUT + co0)) * OHW + oh) * OHW + owb;
            float4 v0, v1;
            v0.x = LREL(a[0] * dm + nv[0] + bias);
            v0.y = LREL(a[1] * dm + nv[1] + bias);
            v0.z = LREL(a[2] * dm + nv[2] + bias);
            v0.w = LREL(a[3] * dm + nv[3] + bias);
            v1.x = LREL(a[4] * dm + nv[4] + bias);
            v1.y = LREL(a[5] * dm + nv[5] + bias);
            v1.z = LREL(a[6] * dm + nv[6] + bias);
            v1.w = LREL(a[7] * dm + nv[7] + bias);
            *(float4*)(op)     = v0;
            *(float4*)(op + 4) = v1;
        }
    }
#undef LREL
}

// ---------------------------------------------------------------------------
extern "C" void kernel_launch(void* const* d_in, const int* in_sizes, int n_in,
                              void* d_out, int out_size, void* d_ws, size_t ws_size,
                              hipStream_t stream) {
    const float* x        = (const float*)d_in[0];
    const float* style    = (const float*)d_in[1];
    const float* noise    = (const float*)d_in[2];
    const float* w_mod    = (const float*)d_in[3];
    const float* b_mod    = (const float*)d_in[4];
    const float* conv_w   = (const float*)d_in[5];
    const float* nwp      = (const float*)d_in[6];
    const float* act_bias = (const float*)d_in[7];
    float* out = (float*)d_out;

    char* ws = (char*)d_ws;
    float* s     = (float*)(ws + 0);                       //  8KB  [8,256]
    float* demod = (float*)(ws + 8192);                    // 16KB  [8,512]
    bf16*  wt    = (bf16*)(ws + 24576);                    // 2.36MB [9,512,256]
    bf16*  xmod  = (bf16*)(ws + 24576 + 2359296);          // 17.3MB [8,65,65,256]

    style_mod_kernel<<<BATCH, 256, 0, stream>>>(style, w_mod, b_mod, s);
    demod_kernel<<<dim3(COUT, BATCH), 256, 0, stream>>>(conv_w, s, demod);
    pack_w_kernel<<<(COUT * CIN) / 256, 256, 0, stream>>>(conv_w, wt);
    fir_mod_pack_kernel<<<BATCH * HP * 8, 256, 0, stream>>>(x, s, xmod);
    modconv_wmma_kernel<<<dim3(64, 4), 256, 0, stream>>>(
        xmod, wt, demod, noise, nwp, act_bias, out);
}
